// STPN_44126493999325
// MI455X (gfx1250) — matrix operand order repaired and checked by reading, not compiled, not run on hardware
//
#include <hip/hip_runtime.h>
#include <cstdint>
#include <cstddef>

// ---------------------------------------------------------------------------
// STPN (graph-LSTM) for MI455X / gfx1250.
// All GEMMs via v_wmma_f32_16x16x32_f16 (f16 operands, f32 accumulation).
// Constant operands pre-packed into fragment-major layout (2x b128 per frag);
// dynamic B operands staged through LDS (ds_load_b128 frag reads).
// Each wave computes 32m x 64n (8 WMMA / K-step, B frags reused 2x).
// wave32, 8 waves / 256-thread block.
// ---------------------------------------------------------------------------

typedef __attribute__((ext_vector_type(16))) _Float16 v16h;
typedef __attribute__((ext_vector_type(8)))  float    v8f;

#define DEVI __device__ __forceinline__
#define WMMA_F16(acc, a, b)                                                    \
  acc = __builtin_amdgcn_wmma_f32_16x16x32_f16(false, (a), false, (b),         \
                                               (short)0, (acc), false, false)

namespace {
constexpr int  Bn = 32, Vn = 500, Tn = 24, VP = 512, Hn = 64, CINn = 2, WEMBn = 4, C0n = 6;
constexpr long NBV  = (long)Bn * VP;     // 16384 columns of the (b,v) space
constexpr long NTBV = (long)Tn * NBV;    // 393216 columns of the (t,b,v) space
}

DEVI float sigm(float x) { return 1.0f / (1.0f + expf(-x)); }

// A-fragment K index for (lane, element): contiguous k = 16*(lane/16) + e, same order as B.
DEVI int akmap(int l, int e) {
  return ((l >> 4) << 4) + e;
}

// ------------------------------- utility -----------------------------------
__global__ void k_zero(float4* __restrict__ p, long n4) {
  long i  = (long)blockIdx.x * blockDim.x + threadIdx.x;
  long st = (long)gridDim.x * blockDim.x;
  float4 z; z.x = z.y = z.z = z.w = 0.f;
  for (; i < n4; i += st) p[i] = z;
}

// Pack f32 (R x C) -> f16 fragment-major A layout, padded (Mp x Kp), batch z.
__global__ void k_packA(const float* __restrict__ src, _Float16* __restrict__ dst,
                        int R, int C, int Mp, int Kp, long srcB, long dstB) {
  const float* s = src + (long)blockIdx.z * srcB;
  _Float16*    d = dst + (long)blockIdx.z * dstB;
  int ktiles = Kp >> 5;
  long n = (long)Mp * Kp;
  for (long i = (long)blockIdx.x * blockDim.x + threadIdx.x; i < n;
       i += (long)gridDim.x * blockDim.x) {
    long tile = i >> 9; int idx = (int)(i & 511);
    int l = idx >> 4, e = idx & 15;
    int mt = (int)(tile / ktiles), kt = (int)(tile % ktiles);
    int m = mt * 16 + (l & 15);
    int k = kt * 32 + akmap(l, e);
    float v = (m < R && k < C) ? s[(long)m * C + k] : 0.f;
    d[i] = (_Float16)v;
  }
}

// Pack f32 (R x C) -> f16 fragment-major B layout, padded (Kp x Np), batch z.
__global__ void k_packB(const float* __restrict__ src, _Float16* __restrict__ dst,
                        int R, int C, int Kp, int Np, long srcB, long dstB) {
  const float* s = src + (long)blockIdx.z * srcB;
  _Float16*    d = dst + (long)blockIdx.z * dstB;
  int ntiles = Np >> 4;
  long n = (long)Kp * Np;
  for (long i = (long)blockIdx.x * blockDim.x + threadIdx.x; i < n;
       i += (long)gridDim.x * blockDim.x) {
    long tile = i >> 9; int idx = (int)(i & 511);
    int l = idx >> 4, e = idx & 15;
    int kt = (int)(tile / ntiles), nt = (int)(tile % ntiles);
    int k  = kt * 32 + ((l >> 4) << 4) + e;
    int nn = nt * 16 + (l & 15);
    float v = (k < R && nn < C) ? s[(long)k * C + nn] : 0.f;
    d[i] = (_Float16)v;
  }
}

// Build layer-0 input L[c][t][b][v] = concat(x, Wemb[w_type]) ; c<6
__global__ void k_embed(const float* __restrict__ x, const int* __restrict__ wt,
                        const float* __restrict__ Wemb, float* __restrict__ L) {
  long n = (long)C0n * Tn * Bn * Vn;
  for (long i = (long)blockIdx.x * blockDim.x + threadIdx.x; i < n;
       i += (long)gridDim.x * blockDim.x) {
    int v = (int)(i % Vn); long r = i / Vn;
    int b = (int)(r % Bn); r /= Bn;
    int t = (int)(r % Tn); int c = (int)(r / Tn);
    float val;
    if (c < CINn) {
      val = x[(((long)b * CINn + c) * Vn + v) * Tn + t];
    } else {
      int w = wt[((long)b * Vn + v) * Tn + t];
      val = Wemb[(long)w * WEMBn + (c - CINn)];
    }
    L[((long)c * Tn + t) * NBV + (long)b * VP + v] = val;
  }
}

// comb = x_t + h_{t-1}  -> f16 into G rows [0, C*B)
__global__ void k_comb(const float* __restrict__ L, const float* __restrict__ Hst,
                       _Float16* __restrict__ G, int C, int tsrc) {
  long n = (long)C * Bn * VP;
  for (long i = (long)blockIdx.x * blockDim.x + threadIdx.x; i < n;
       i += (long)gridDim.x * blockDim.x) {
    int v = (int)(i % VP); long r = i / VP;         // r = c*B + b
    int b = (int)(r % Bn); int c = (int)(r / Bn);
    float val = L[((long)c * Tn + tsrc) * NBV + (long)b * VP + v] + Hst[r * VP + v];
    G[r * VP + v] = (_Float16)val;
  }
}

// ---------------------------------------------------------------------------
// Diffusion GEMM: D(f16) = Asrc(row-major f16, dynamic) x Bpk(packed frags).
// N fixed = 512 (VP): 8 waves each own a 64-wide strip; a 32-row A tile is
// staged per K-chunk in LDS fragment layout, shared by all waves; each wave
// runs 8 WMMA per K-step reusing every B fragment twice.
// grid.x = M/32, grid.z = 3 supports.
// ---------------------------------------------------------------------------
__global__ void __launch_bounds__(256)
k_diff(const _Float16* __restrict__ Asrc, long lda, long aR0, long aRZ,
       const _Float16* __restrict__ Bpk, long bZ,
       _Float16* __restrict__ D, long ldd, long dR0, long dRZ, int K) {
  __shared__ _Float16 sA[1024];
  int z = blockIdx.z;
  const _Float16* As = Asrc + (aR0 + (long)z * aRZ) * lda;
  const _Float16* Bp = Bpk + (long)z * bZ;
  _Float16* Dst = D + (dR0 + (long)z * dRZ) * ldd;
  long m0 = (long)blockIdx.x * 32;
  int tid = threadIdx.x, lane = tid & 31, wv = tid >> 5;
  v8f a00 = {}, a01 = {}, a02 = {}, a03 = {};
  v8f a10 = {}, a11 = {}, a12 = {}, a13 = {};
  int ktiles = K >> 5;
  int sl = tid >> 3, se0 = (tid & 7) << 1;
  for (int kt = 0; kt < ktiles; ++kt) {
    {  // stage A (32x32) into fragment layout: 2x b32 per thread
      int k = (kt << 5) + akmap(sl, se0);
      unsigned u0 = *(const unsigned*)(As + (m0 + (sl & 15)) * lda + k);
      unsigned u1 = *(const unsigned*)(As + (m0 + 16 + (sl & 15)) * lda + k);
      *(unsigned*)(sA + sl * 16 + se0) = u0;
      *(unsigned*)(sA + 512 + sl * 16 + se0) = u1;
    }
    __syncthreads();
    v16h af0 = *(const v16h*)(sA + lane * 16);
    v16h af1 = *(const v16h*)(sA + 512 + lane * 16);
    const _Float16* bb = Bp + (((long)kt << 5) + (wv << 2)) * 512 + lane * 16;
    v16h f0 = *(const v16h*)(bb);
    v16h f1 = *(const v16h*)(bb + 512);
    v16h f2 = *(const v16h*)(bb + 1024);
    v16h f3 = *(const v16h*)(bb + 1536);
    WMMA_F16(a00, af0, f0); WMMA_F16(a10, af1, f0);
    WMMA_F16(a01, af0, f1); WMMA_F16(a11, af1, f1);
    WMMA_F16(a02, af0, f2); WMMA_F16(a12, af1, f2);
    WMMA_F16(a03, af0, f3); WMMA_F16(a13, af1, f3);
    __syncthreads();
  }
  long n0 = (long)wv * 64;
  int g = lane >> 4, nn = lane & 15;
#pragma unroll
  for (int e = 0; e < 8; ++e) {
    _Float16* dp = Dst + (m0 + e + 8 * g) * ldd + n0 + nn;
    dp[0]  = (_Float16)a00[e];
    dp[16] = (_Float16)a01[e];
    dp[32] = (_Float16)a02[e];
    dp[48] = (_Float16)a03[e];
    _Float16* dq = dp + 16 * ldd;
    dq[0]  = (_Float16)a10[e];
    dq[16] = (_Float16)a11[e];
    dq[32] = (_Float16)a12[e];
    dq[48] = (_Float16)a13[e];
  }
}

// ---------------------------------------------------------------------------
// GEMM f32-out: D = Ap(packed frags) x Bsrc(row-major f16, dynamic) (+bias).
// Block = 8 waves (4 m-waves x 2 n-halves); each m-wave owns two strided
// m-tiles (mt, mt+4) -> 128m x 128n per block.  The 32x128 K-slab of B is
// staged transposed in LDS (k-pair packed) so a fragment is two contiguous
// ds_load_b128 per lane, reused by 8 WMMAs.
// grid.x = N/128, grid.y = ceil(M/128).
// ---------------------------------------------------------------------------
__global__ void __launch_bounds__(256)
k_gemm_ldsb(const _Float16* __restrict__ Ap,
            const _Float16* __restrict__ Bsrc, long ldb,
            float* __restrict__ D, long ldd,
            int M, int K, const float* __restrict__ bias, int biasN) {
  __shared__ _Float16 sB[128 * 32];   // [n][k], k-pairs packed as b32
  int tid = threadIdx.x, lane = tid & 31, wv = tid >> 5;
  int wm = wv & 3, wn = wv >> 2;
  int mtA = blockIdx.y * 8 + wm;
  int mtB = mtA + 4;
  bool actA = (mtA << 4) < M;
  bool actB = (mtB << 4) < M;
  long n0 = (long)blockIdx.x * 128;
  int ktiles = K >> 5;
  v8f a00 = {}, a01 = {}, a02 = {}, a03 = {};
  v8f a10 = {}, a11 = {}, a12 = {}, a13 = {};
  for (int kt = 0; kt < ktiles; ++kt) {
    {  // stage: thread loads 8 halves from rows 2kp and 2kp+1, packs k-pairs
      int kp = tid >> 4;                 // 0..15
      int nn = (tid & 15) << 3;          // 0..120
      const _Float16* s0 = Bsrc + ((long)(kt << 5) + 2 * kp) * ldb + n0 + nn;
      const _Float16* s1 = s0 + ldb;
      uint4 u0 = *(const uint4*)s0;
      uint4 u1 = *(const uint4*)s1;
      const unsigned short* h0 = (const unsigned short*)&u0;
      const unsigned short* h1 = (const unsigned short*)&u1;
      unsigned* sB32 = (unsigned*)sB;
#pragma unroll
      for (int j = 0; j < 8; ++j)
        sB32[(nn + j) * 16 + kp] = (unsigned)h0[j] | ((unsigned)h1[j] << 16);
      if (kt + 1 < ktiles)               // global_prefetch_b8 next K-slab
        __builtin_prefetch((const void*)(s0 + 32 * ldb), 0, 1);
    }
    __syncthreads();
    if (actA) {
      const _Float16* bb = sB + ((wn << 6) + (lane & 15)) * 32 + ((lane >> 4) << 4);
      v16h f0 = *(const v16h*)(bb);
      v16h f1 = *(const v16h*)(bb + 16 * 32);
      v16h f2 = *(const v16h*)(bb + 32 * 32);
      v16h f3 = *(const v16h*)(bb + 48 * 32);
      v16h afA = *(const v16h*)(Ap + ((long)mtA * ktiles + kt) * 512 + lane * 16);
      WMMA_F16(a00, afA, f0);
      WMMA_F16(a01, afA, f1);
      WMMA_F16(a02, afA, f2);
      WMMA_F16(a03, afA, f3);
      if (actB) {
        v16h afB = *(const v16h*)(Ap + ((long)mtB * ktiles + kt) * 512 + lane * 16);
        WMMA_F16(a10, afB, f0);
        WMMA_F16(a11, afB, f1);
        WMMA_F16(a12, afB, f2);
        WMMA_F16(a13, afB, f3);
      }
    }
    __syncthreads();
  }
  long nb = n0 + (long)(wn << 6);
  int g = lane >> 4, nn = lane & 15;
  if (actA) {
#pragma unroll
    for (int e = 0; e < 8; ++e) {
      long mrow = (long)mtA * 16 + e + 8 * g;
      float bv = (bias != nullptr && mrow < biasN) ? bias[mrow] : 0.f;
      float* dp = D + mrow * ldd + nb + nn;
      dp[0]  = a00[e] + bv;
      dp[16] = a01[e] + bv;
      dp[32] = a02[e] + bv;
      dp[48] = a03[e] + bv;
    }
  }
  if (actB) {
#pragma unroll
    for (int e = 0; e < 8; ++e) {
      long mrow = (long)mtB * 16 + e + 8 * g;
      float bv = (bias != nullptr && mrow < biasN) ? bias[mrow] : 0.f;
      float* dp = D + mrow * ldd + nb + nn;
      dp[0]  = a10[e] + bv;
      dp[16] = a11[e] + bv;
      dp[32] = a12[e] + bv;
      dp[48] = a13[e] + bv;
    }
  }
}

// LSTM pointwise: cy = sig(f)*cx + sig(i)*cg ; hy = sig(o)*tanh(cy)
__global__ void k_lstm(const float* __restrict__ gates, const float* __restrict__ bg,
                       float* __restrict__ Cst, float* __restrict__ Hst,
                       _Float16* __restrict__ HS, int C, int t) {
  long n = (long)C * Bn * Vn;
  for (long i = (long)blockIdx.x * blockDim.x + threadIdx.x; i < n;
       i += (long)gridDim.x * blockDim.x) {
    int v = (int)(i % Vn); long r = i / Vn;
    int b = (int)(r % Bn); int c = (int)(r / Bn);
    long col = (long)b * VP + v;
    float gi = gates[((long)(0 * C + c)) * NBV + col] + bg[0 * C + c];
    float gf = gates[((long)(1 * C + c)) * NBV + col] + bg[1 * C + c];
    float gc = gates[((long)(2 * C + c)) * NBV + col] + bg[2 * C + c];
    float go = gates[((long)(3 * C + c)) * NBV + col] + bg[3 * C + c];
    long sidx = ((long)c * Bn + b) * VP + v;
    float cy = sigm(gf) * Cst[sidx] + sigm(gi) * gc;   // linear cell-gate (no tanh)
    float hy = sigm(go) * tanhf(cy);
    Cst[sidx] = cy;
    Hst[sidx] = hy;
    HS[((long)c * Tn + t) * NBV + col] = (_Float16)hy;
  }
}

// ------------------------------- SE block ----------------------------------
__global__ void k_se_reduce(const float* __restrict__ L, float* __restrict__ y) {
  int c = blockIdx.x / Bn, b = blockIdx.x % Bn;
  float s = 0.f;
  for (long i = threadIdx.x; i < (long)Tn * Vn; i += blockDim.x) {
    int t = (int)(i / Vn), v = (int)(i % Vn);
    s += L[((long)c * Tn + t) * NBV + (long)b * VP + v];
  }
  __shared__ float sm[256];
  sm[threadIdx.x] = s; __syncthreads();
  for (int o = 128; o > 0; o >>= 1) {
    if (threadIdx.x < o) sm[threadIdx.x] += sm[threadIdx.x + o];
    __syncthreads();
  }
  if (threadIdx.x == 0) y[(long)b * Hn + c] = sm[0] / (float)(Tn * Vn);
}

__global__ void k_se_mlp(const float* __restrict__ y, const float* __restrict__ Wa,
                         const float* __restrict__ Wb, float* __restrict__ sc) {
  int i = blockIdx.x * blockDim.x + threadIdx.x;
  if (i >= Bn * Hn) return;
  int b = i / Hn, c = i % Hn;
  float a[4];
#pragma unroll
  for (int j = 0; j < 4; ++j) {
    float s = 0.f;
    for (int k = 0; k < Hn; ++k) s += y[(long)b * Hn + k] * Wa[(long)j * Hn + k];
    a[j] = fmaxf(s, 0.f);
  }
  float s = 0.f;
#pragma unroll
  for (int j = 0; j < 4; ++j) s += a[j] * Wb[(long)c * 4 + j];
  sc[i] = sigm(s);
}

__global__ void k_se_scale(float* __restrict__ L, const float* __restrict__ sc) {
  long n = (long)Hn * Tn * NBV;
  for (long i = (long)blockIdx.x * blockDim.x + threadIdx.x; i < n;
       i += (long)gridDim.x * blockDim.x) {
    long r = i / VP;
    int b = (int)(r % Bn); r /= Bn;
    r /= Tn;                       // skip t
    int c = (int)r;
    L[i] *= sc[(long)b * Hn + c];
  }
}

// final: d_out[b][o][v][t] = L[o][t][b][v], o<2, v<V
__global__ void k_out(const float* __restrict__ L, float* __restrict__ out) {
  long n = (long)Bn * 2 * Vn * Tn;
  for (long i = (long)blockIdx.x * blockDim.x + threadIdx.x; i < n;
       i += (long)gridDim.x * blockDim.x) {
    int t = (int)(i % Tn); long r = i / Tn;
    int v = (int)(r % Vn); r /= Vn;
    int o = (int)(r % 2);  int b = (int)(r / 2);
    out[i] = L[((long)o * Tn + t) * NBV + (long)b * VP + v];
  }
}

// ------------------------------- host side ---------------------------------
static inline unsigned gblocks(long n, int tpb = 256, long cap = 8192) {
  long b = (n + tpb - 1) / tpb;
  if (b > cap) b = cap;
  if (b < 1) b = 1;
  return (unsigned)b;
}

extern "C" void kernel_launch(void* const* d_in, const int* in_sizes, int n_in,
                              void* d_out, int out_size, void* d_ws, size_t ws_size,
                              hipStream_t stream) {
  (void)in_sizes; (void)n_in; (void)out_size; (void)ws_size;

  const float* x    = (const float*)d_in[0];
  const float* sup  = (const float*)d_in[1];
  const int*   wt   = (const int*)d_in[2];
  const float* Wemb = (const float*)d_in[3];
  const float* Wg0 = (const float*)d_in[4];  const float* bg0 = (const float*)d_in[5];
  const float* Wo0 = (const float*)d_in[6];  const float* bo0 = (const float*)d_in[7];
  const float* Wa0 = (const float*)d_in[8];  const float* Wb0 = (const float*)d_in[9];
  const float* Wg1 = (const float*)d_in[10]; const float* bg1 = (const float*)d_in[11];
  const float* Wo1 = (const float*)d_in[12]; const float* bo1 = (const float*)d_in[13];
  const float* Wa1 = (const float*)d_in[14]; const float* Wb1 = (const float*)d_in[15];
  const float* Wg2 = (const float*)d_in[16]; const float* bg2 = (const float*)d_in[17];
  const float* Wo2 = (const float*)d_in[18]; const float* bo2 = (const float*)d_in[19];
  const float* Wg3 = (const float*)d_in[20]; const float* bg3 = (const float*)d_in[21];
  const float* Wo3 = (const float*)d_in[22]; const float* bo3 = (const float*)d_in[23];
  float* out = (float*)d_out;

  // ---- workspace carve-out (all 256B aligned) ----
  char* ws = (char*)d_ws;
  size_t off = 0;
  auto take = [&](size_t bytes) -> void* {
    off = (off + 255) & ~(size_t)255;
    void* p = ws + off;
    off += bytes;
    return p;
  };
  float*    L     = (float*)   take((size_t)Hn * Tn * NBV * 4);   // (64,T,B,VP) f32
  _Float16* HS    = (_Float16*)take((size_t)Hn * Tn * NBV * 2);   // (64,T,B,VP) f16
  _Float16* G     = (_Float16*)take((size_t)7 * Hn * NBV * 2);    // (448,B,VP)  f16
  float*    Gates = (float*)   take((size_t)4 * Hn * NBV * 4);    // (256,B,VP)  f32
  float*    Hst   = (float*)   take((size_t)Hn * NBV * 4);
  float*    Cst   = (float*)   take((size_t)Hn * NBV * 4);
  _Float16* AhPk  = (_Float16*)take((size_t)3 * VP * VP * 2);     // packed B frags
  _Float16* Wg0p  = (_Float16*)take((size_t)32 * 64 * 2);         // packed A frags
  _Float16* Wg1p  = (_Float16*)take((size_t)256 * 448 * 2);
  _Float16* Wg2p  = (_Float16*)take((size_t)256 * 448 * 2);
  _Float16* Wg3p  = (_Float16*)take((size_t)256 * 448 * 2);
  _Float16* Wo0p  = (_Float16*)take((size_t)64 * 32 * 2);
  _Float16* Wo1p  = (_Float16*)take((size_t)64 * 64 * 2);
  _Float16* Wo2p  = (_Float16*)take((size_t)64 * 64 * 2);
  _Float16* Wo3p  = (_Float16*)take((size_t)16 * 64 * 2);
  float*    yb    = (float*)   take((size_t)Bn * Hn * 4);
  float*    sc    = (float*)   take((size_t)Bn * Hn * 4);
  size_t total = (off + 255) & ~(size_t)255;

  // ---- zero entire scratch (pads must be 0) ----
  k_zero<<<gblocks((long)(total / 16)), 256, 0, stream>>>((float4*)ws, (long)(total / 16));

  // ---- weight prep: convert + pad + fragment-pack ----
  k_packB<<<dim3(gblocks((long)VP * VP), 1, 3), 256, 0, stream>>>(
      sup, AhPk, Vn, Vn, VP, VP, (long)Vn * Vn, (long)VP * VP);
  k_packA<<<dim3(gblocks(32L * 64)), 256, 0, stream>>>(Wg0, Wg0p, 24, 42, 32, 64, 0, 0);
  k_packA<<<dim3(gblocks(256L * 448)), 256, 0, stream>>>(Wg1, Wg1p, 256, 448, 256, 448, 0, 0);
  k_packA<<<dim3(gblocks(256L * 448)), 256, 0, stream>>>(Wg2, Wg2p, 256, 448, 256, 448, 0, 0);
  k_packA<<<dim3(gblocks(256L * 448)), 256, 0, stream>>>(Wg3, Wg3p, 256, 448, 256, 448, 0, 0);
  k_packA<<<dim3(gblocks(64L * 32)), 256, 0, stream>>>(Wo0, Wo0p, 64, 6, 64, 32, 0, 0);
  k_packA<<<dim3(gblocks(64L * 64)), 256, 0, stream>>>(Wo1, Wo1p, 64, 64, 64, 64, 0, 0);
  k_packA<<<dim3(gblocks(64L * 64)), 256, 0, stream>>>(Wo2, Wo2p, 64, 64, 64, 64, 0, 0);
  k_packA<<<dim3(gblocks(16L * 64)), 256, 0, stream>>>(Wo3, Wo3p, 2, 64, 16, 64, 0, 0);

  // ---- layer-0 input ----
  k_embed<<<gblocks((long)C0n * Tn * Bn * Vn), 256, 0, stream>>>(x, wt, Wemb, L);

  // One ST-G-LSTM layer: scan over T with WMMA diffusion + gate GEMMs,
  // then one big WMMA output-projection GEMM over all (t,b,v) columns.
  auto run_layer = [&](const _Float16* Wgp, const float* bg,
                       const _Float16* Wop, const float* bo,
                       int C, int MpG, int KpG, int MpP, int KpP, int biasN, bool fin) {
    long CB = (long)C * Bn;
    k_zero<<<gblocks(CB * VP / 4), 256, 0, stream>>>((float4*)Hst, CB * VP / 4);
    k_zero<<<gblocks(CB * VP / 4), 256, 0, stream>>>((float4*)Cst, CB * VP / 4);
    for (int t = 0; t < Tn; ++t) {
      int ts = fin ? (Tn - 1) : t;
      k_comb<<<gblocks(CB * VP), 256, 0, stream>>>(L, Hst, G, C, ts);
      // x1_s = comb * A_s   (rows (1+2s)*CB), z over 3 supports
      k_diff<<<dim3((unsigned)(CB / 32), 1, 3), 256, 0, stream>>>(
          G, VP, 0, 0, AhPk, (long)VP * VP, G, VP, CB, 2 * CB, VP);
      // x2_s = x1_s * A_s   (rows (2+2s)*CB)
      k_diff<<<dim3((unsigned)(CB / 32), 1, 3), 256, 0, stream>>>(
          G, VP, CB, 2 * CB, AhPk, (long)VP * VP, G, VP, 2 * CB, 2 * CB, VP);
      // gates = Wg * feats  over N = B*VP columns
      k_gemm_ldsb<<<dim3((unsigned)(NBV / 128), (unsigned)((MpG + 127) / 128)), 256, 0, stream>>>(
          Wgp, G, NBV, Gates, NBV, MpG, KpG, nullptr, 0);
      k_lstm<<<gblocks((long)C * Bn * Vn), 256, 0, stream>>>(Gates, bg, Cst, Hst, HS, C, t);
    }
    // out = Wo * hs + bo over N = T*B*VP columns (overwrites L in place)
    k_gemm_ldsb<<<dim3((unsigned)(NTBV / 128), (unsigned)((MpP + 127) / 128)), 256, 0, stream>>>(
        Wop, HS, NTBV, L, NTBV, MpP, KpP, bo, biasN);
  };

  auto run_se = [&](const float* Wa, const float* Wb) {
    k_se_reduce<<<dim3(Hn * Bn), 256, 0, stream>>>(L, yb);
    k_se_mlp<<<dim3((Bn * Hn + 255) / 256), 256, 0, stream>>>(yb, Wa, Wb, sc);
    k_se_scale<<<gblocks((long)Hn * Tn * NBV), 256, 0, stream>>>(L, sc);
  };

  //                 Wg    bg   Wo    bo   C   MpG  KpG  MpP KpP biasN final
  run_layer(Wg0p, bg0, Wo0p, bo0, C0n, 32, 64, 64, 32, 64, false);
  run_se(Wa0, Wb0);
  run_layer(Wg1p, bg1, Wo1p, bo1, Hn, 256, 448, 64, 64, 64, false);
  run_se(Wa1, Wb1);
  run_layer(Wg2p, bg2, Wo2p, bo2, Hn, 256, 448, 64, 64, 64, false);
  run_layer(Wg3p, bg3, Wo3p, bo3, Hn, 256, 448, 16, 64, 2, true);

  k_out<<<gblocks((long)Bn * 2 * Vn * Tn), 256, 0, stream>>>(L, out);
}
